// FP16ToFP32Converter_43860206027266
// MI455X (gfx1250) — compile-verified, matrix operand order checked by reading, not run on hardware
//
#include <hip/hip_runtime.h>

typedef __attribute__((ext_vector_type(16))) _Float16 v16h;
typedef __attribute__((ext_vector_type(8)))  float    v8f;
typedef __attribute__((ext_vector_type(4)))  float    f4;

// Bit-exact FP16 -> FP32 conversion on the packed 16-bit pattern.
// Fully branchless: every select is mask arithmetic (mask = 0 - cond),
// so no exec-mask regions are generated.
__device__ __forceinline__ unsigned fp16_bits_to_fp32_bits(unsigned pu) {
  const unsigned s = (pu >> 15) & 1u;
  const unsigned e = (pu >> 10) & 31u;
  const unsigned m = pu & 1023u;

  // Subnormal candidate: normalize the 10-bit mantissa.
  const unsigned p        = 31u - (unsigned)__clz((int)(m | 1u)); // msb pos 0..9
  const unsigned sub_mant = ((m << (10u - p)) & 1023u) << 13;     // 0 when m==0

  // Masks (all-ones when condition true).
  const unsigned nz_mask   = 0u - (unsigned)(m != 0u);
  const unsigned emax_mask = 0u - (unsigned)(e == 31u);
  const unsigned e0_mask   = 0u - (unsigned)(e == 0u);

  // Normal / inf / NaN exponent: e+112 == 143 for e==31; 143 | 0x70 == 255.
  const unsigned exp_norm = (e + 112u) | (emax_mask & 0x70u);
  // Subnormal exponent; forced to 0 for true zero (m==0).
  const unsigned exp_sub  = (103u + p) & nz_mask;

  const unsigned exp32  = (exp_sub  & e0_mask) | (exp_norm  & ~e0_mask);
  const unsigned mant23 = (sub_mant & e0_mask) | ((m << 13) & ~e0_mask);

  return (s << 31) | (exp32 << 23) | mant23;
}

// Unpack the 32-bit pattern into 32 floats (0.0/1.0), MSB first, NT stores.
__device__ __forceinline__ void unpack_and_store(unsigned w, float* __restrict__ orow) {
  f4* o4 = (f4*)orow;
#pragma unroll
  for (int i = 0; i < 8; ++i) {
    f4 v;
    v.x = (float)((w >> (31 - (4 * i + 0))) & 1u);
    v.y = (float)((w >> (31 - (4 * i + 1))) & 1u);
    v.z = (float)((w >> (31 - (4 * i + 2))) & 1u);
    v.w = (float)((w >> (31 - (4 * i + 3))) & 1u);
    __builtin_nontemporal_store(v, o4 + i);   // global_store_b128, TH=NT
  }
}

// One wave (32 lanes) handles 32 rows: two 16x16-bit tiles packed via
// v_wmma_f32_16x16x32_f16. Weights are folded into A (every product is a
// power of two <= 2^15, exact in f16; accumulation exact in f32), B = ones,
// so D[M,N] = packed[M] independent of B's K-striping details.
__global__ void __launch_bounds__(256)
fp16bits_to_fp32bits_kernel(const float* __restrict__ in, float* __restrict__ out, int rows) {
  const int lane = threadIdx.x & 31;
  const int wave = threadIdx.x >> 5;
  const int R0   = blockIdx.x * 256 + wave * 32;

  if (R0 + 32 <= rows) {
    // --- WMMA pack path (whole wave active: EXEC all ones) ---
    const int   halfsel = lane >> 4;        // 0: bits 0..7 of row, 1: bits 8..15
    const int   mrow    = lane & 15;        // M within the 16-row tile
    const float wscale  = halfsel ? 1.0f : 256.0f;

    v16h b;
#pragma unroll
    for (int i = 0; i < 16; ++i) b[i] = (_Float16)1.0f;
    v8f c = {};
    v8f d0, d1;

#pragma unroll
    for (int t = 0; t < 2; ++t) {
      const float* rowp = in + (size_t)(R0 + t * 16 + mrow) * 16 + (size_t)halfsel * 8;
      const f4 q0 = __builtin_nontemporal_load((const f4*)rowp);      // TH=NT
      const f4 q1 = __builtin_nontemporal_load(((const f4*)rowp) + 1);
      v16h a = {};                            // elements 8..15 (K=16..31) stay zero
      a[0] = (_Float16)(q0.x * (wscale * 128.0f));
      a[1] = (_Float16)(q0.y * (wscale *  64.0f));
      a[2] = (_Float16)(q0.z * (wscale *  32.0f));
      a[3] = (_Float16)(q0.w * (wscale *  16.0f));
      a[4] = (_Float16)(q1.x * (wscale *   8.0f));
      a[5] = (_Float16)(q1.y * (wscale *   4.0f));
      a[6] = (_Float16)(q1.z * (wscale *   2.0f));
      a[7] = (_Float16)(q1.w * (wscale *   1.0f));
      v8f d = __builtin_amdgcn_wmma_f32_16x16x32_f16(
          /*neg_a=*/false, a, /*neg_b=*/false, b,
          /*c_mod=*/(short)0, c, /*reuse_a=*/false, /*reuse_b=*/false);
      if (t == 0) d0 = d; else d1 = d;
    }

    // Lane -> unique row mapping. D layout: VGPR v, lanes 0-15 hold M=v,
    // lanes 16-31 hold M=v+8 (value replicated across N since B is uniform).
    const int j  = lane & 7;
    const int hi = (lane >> 3) & 1;            // which 16-row tile
    v8f ds = hi ? d1 : d0;
    float s01 = (j & 1) ? ds[1] : ds[0];
    float s23 = (j & 1) ? ds[3] : ds[2];
    float s45 = (j & 1) ? ds[5] : ds[4];
    float s67 = (j & 1) ? ds[7] : ds[6];
    float s03 = (j & 2) ? s23 : s01;
    float s47 = (j & 2) ? s67 : s45;
    float pv  = (j & 4) ? s47 : s03;           // exact integer 0..65535

    const unsigned pu = (unsigned)pv;
    const unsigned w  = fp16_bits_to_fp32_bits(pu);
    const int r = R0 + halfsel * 8 + hi * 16 + j;   // bijection onto R0..R0+31
    unpack_and_store(w, out + (size_t)r * 32);
  } else {
    // --- scalar tail (partial wave; never taken for rows % 256 == 0) ---
    const int r = R0 + lane;
    if (r < rows) {
      const float* rowp = in + (size_t)r * 16;
      unsigned pu = 0u;
#pragma unroll
      for (int i = 0; i < 16; ++i)
        pu |= (rowp[i] > 0.5f ? 1u : 0u) << (15 - i);
      unpack_and_store(fp16_bits_to_fp32_bits(pu), out + (size_t)r * 32);
    }
  }
}

extern "C" void kernel_launch(void* const* d_in, const int* in_sizes, int n_in,
                              void* d_out, int out_size, void* d_ws, size_t ws_size,
                              hipStream_t stream) {
  (void)n_in; (void)out_size; (void)d_ws; (void)ws_size;
  const float* in  = (const float*)d_in[0];
  float*       out = (float*)d_out;
  const int rows   = in_sizes[0] / 16;          // N = 4,194,304
  const int blocks = (rows + 255) / 256;        // 256 threads = 8 waves = 256 rows/block
  fp16bits_to_fp32bits_kernel<<<dim3(blocks), dim3(256), 0, stream>>>(in, out, rows);
}
